// LRTM_35862976922232
// MI455X (gfx1250) — compile-verified, matrix operand order checked by reading, not run on hardware
//
#include <hip/hip_runtime.h>
#include <hip/hip_bf16.h>

// ---------------------------------------------------------------------------
// MI455X (gfx1250, wave32) implementation of the 2-layer transformer block.
// All matmuls (QKV, out-proj, FC1, FC2, Q*K^T, P*V) run on v_wmma_f32_16x16x32_bf16
// (bf16 inputs, f32 accumulate). GEMMs stage the weight panel in LDS via the
// gfx1250 async-to-LDS path (ASYNCcnt); attention keeps the 16x1024 score
// strip in LDS.
// ---------------------------------------------------------------------------

typedef unsigned short ushort_t;
typedef __attribute__((ext_vector_type(16))) __bf16 v16bf;
typedef __attribute__((ext_vector_type(8)))  __bf16 v8bf;
typedef __attribute__((ext_vector_type(8)))  float  v8f;
typedef __attribute__((ext_vector_type(4)))  int    v4i;

#define LNUM 2
#define BB   8
#define NN   1024
#define DD   256
#define HH   4
#define DHD  64
#define MMF  512
#define EPSV 1e-5f

// ---------------- helpers ----------------
__device__ __forceinline__ ushort_t f2bf(float f) {
  union { float f; unsigned int u; } v; v.f = f;
  unsigned int u = v.u;
  u += 0x7FFFu + ((u >> 16) & 1u);          // round-to-nearest-even
  return (ushort_t)(u >> 16);
}
__device__ __forceinline__ float bf2f(ushort_t h) {
  union { unsigned int u; float f; } v; v.u = ((unsigned int)h) << 16; return v.f;
}
__device__ __forceinline__ __bf16 us2bf(ushort_t u) {
  __bf16 r; __builtin_memcpy(&r, &u, 2); return r;
}
__device__ __forceinline__ float geluf(float x) {
  return 0.5f * x * (1.0f + erff(x * 0.70710678118654752f));
}
__device__ __forceinline__ v8f wmma_bf16(v16bf a, v16bf b, v8f c) {
  return __builtin_amdgcn_wmma_f32_16x16x32_bf16(false, a, false, b, (short)0, c,
                                                 false, false);
}
// 16-bit A/B tile loader from row-major [16 x >=32] @ (base,stride).
// Per ISA layout: lanes 0-15 (row=lane) hold K={0..7,16..23}, lanes 16-31 hold
// K={8..15,24..31}. Two contiguous 16B loads per lane.
__device__ __forceinline__ v16bf ld_tile(const ushort_t* p0, int stride, int lane) {
  int row = lane & 15;
  int hi  = (lane >> 4) & 1;
  const ushort_t* p = p0 + row * stride + hi * 8;
  v8bf lo = *(const v8bf*)(p);
  v8bf hb = *(const v8bf*)(p + 16);
  v16bf r;
#pragma unroll
  for (int i = 0; i < 8; ++i) { r[i] = lo[i]; r[i + 8] = hb[i]; }
  return r;
}
// K index for element i of a gathered (non-contiguous) B tile.
__device__ __forceinline__ int kmapB(int i, int lane) {
  return ((lane >> 4) & 1) * 8 + (i < 8 ? i : i + 8);
}

// ---------------- weight cast + transpose: f32[K][Nout] -> bf16[Nout][K] ----
__global__ void k_castT(const float* __restrict__ src, ushort_t* __restrict__ dst,
                        int K, int Nout, int total) {
  int idx = blockIdx.x * blockDim.x + threadIdx.x;
  if (idx >= total) return;
  int k = idx / Nout, n = idx % Nout;
  dst[(size_t)n * K + k] = f2bf(src[idx]);
}

// ---------------- LayerNorm (per row of D=256) + bf16 cast --------------------
__global__ void k_ln_cast(const float* __restrict__ X, const float* __restrict__ g,
                          const float* __restrict__ bta, ushort_t* __restrict__ H) {
  __shared__ float red[DD];
  int row = blockIdx.x, t = threadIdx.x;
  float v = X[(size_t)row * DD + t];
  red[t] = v; __syncthreads();
  for (int s = DD / 2; s > 0; s >>= 1) { if (t < s) red[t] += red[t + s]; __syncthreads(); }
  float mu = red[0] * (1.0f / DD); __syncthreads();
  float d = v - mu;
  red[t] = d * d; __syncthreads();
  for (int s = DD / 2; s > 0; s >>= 1) { if (t < s) red[t] += red[t + s]; __syncthreads(); }
  float rstd = rsqrtf(red[0] * (1.0f / DD) + EPSV);
  H[(size_t)row * DD + t] = f2bf(d * rstd * g[t] + bta[t]);
}

// ---------------- WMMA GEMM with LDS-staged weight panel ---------------------
// C[M,Nc] = A[M,K](bf16) x Bt[Nc,K](bf16), block = 256 thr (8 waves).
// Block tile: 128 rows x 32 cols. B panel [32 x K] staged in LDS once
// (async-to-LDS), reused by all 8 waves; each wave computes a 16x32 strip
// (A tile reused across 2 WMMAs per k-step).
__global__ __launch_bounds__(256)
void k_gemm(const ushort_t* __restrict__ A, const ushort_t* __restrict__ Bt,
            const float* __restrict__ bias, const float* __restrict__ res,
            float* __restrict__ outF, ushort_t* __restrict__ outBf,
            int K, int Nc, int act) {
  __shared__ __align__(16) ushort_t Bp[32 * 512];   // up to 32KB (K<=512)
  const int t = threadIdx.x, lane = t & 31, w = t >> 5;
  const int n0 = blockIdx.x * 32;
  const int m0 = blockIdx.y * 128 + w * 16;

  // ---- stage B panel [32 x K] into LDS ----
  {
    const int kch = K >> 3;                 // 16B chunks per row
    const int chunks = 32 * kch;
    for (int c = t; c < chunks; c += 256) {
      int row = c / kch;
      int kc  = (c - row * kch) << 3;
      const ushort_t* gp = Bt + (size_t)(n0 + row) * K + kc;
      ushort_t* lp = &Bp[row * K + kc];
#if __has_builtin(__builtin_amdgcn_global_load_async_to_lds_b128)
      __builtin_amdgcn_global_load_async_to_lds_b128(
          (__attribute__((address_space(1))) v4i*)const_cast<ushort_t*>(gp),
          (__attribute__((address_space(3))) v4i*)lp, 0, 0);
#else
      *(v8bf*)lp = *(const v8bf*)gp;
#endif
    }
#if __has_builtin(__builtin_amdgcn_global_load_async_to_lds_b128)
#if __has_builtin(__builtin_amdgcn_s_wait_asynccnt)
    __builtin_amdgcn_s_wait_asynccnt(0);
#else
    asm volatile("s_wait_asynccnt 0" ::: "memory");
#endif
#endif
  }
  __syncthreads();

  // ---- main loop ----
  v8f c0 = {}, c1 = {};
  const ushort_t* arow = A + (size_t)m0 * K;
  for (int k0 = 0; k0 < K; k0 += 32) {
    v16bf a  = ld_tile(arow + k0, K, lane);
    v16bf b0 = ld_tile(&Bp[0]        + k0, K, lane);  // panel rows 0..15
    v16bf b1 = ld_tile(&Bp[16 * K]   + k0, K, lane);  // panel rows 16..31
    c0 = wmma_bf16(a, b0, c0);
    c1 = wmma_bf16(a, b1, c1);
  }

  // ---- epilogue ----
  int col = lane & 15, hi = (lane >> 4) & 1;
#pragma unroll
  for (int tile = 0; tile < 2; ++tile) {
    int nc = n0 + tile * 16 + col;
    float bvv = bias ? bias[nc] : 0.0f;
#pragma unroll
    for (int r = 0; r < 8; ++r) {
      int m = m0 + r + hi * 8;
      float v = (tile ? c1[r] : c0[r]) + bvv;
      size_t oi = (size_t)m * Nc + nc;
      if (res) v += res[oi];
      if (act) v = geluf(v);
      if (outF)  outF[oi]  = v;
      if (outBf) outBf[oi] = f2bf(v);
    }
  }
}

// ---------------- attention: per (b,h,16-row q block); scores live in LDS ----
__global__ __launch_bounds__(128)
void k_attn(const ushort_t* __restrict__ qkv, ushort_t* __restrict__ obf) {
  __shared__ __align__(16) ushort_t Sld[16 * NN];   // 32KB: S, then P in place
  __shared__ float part[16 * 8];
  __shared__ float rowmax[16];
  __shared__ float rowsum[16];

  const int qb = blockIdx.x, h = blockIdx.y, b = blockIdx.z;
  const int t = threadIdx.x, lane = t & 31, w = t >> 5;
  const int m0 = qb * 16;
  const float scale = 0.125f;                      // 64^-0.5

  // Q tile (16 x 64) held in registers, reused across all 64 j tiles.
  const ushort_t* qptr = qkv + ((size_t)(b * NN + m0) * (3 * HH * DHD) + h * DHD);
  v16bf aq0 = ld_tile(qptr,      3 * HH * DHD, lane);
  v16bf aq1 = ld_tile(qptr + 32, 3 * HH * DHD, lane);

  // Phase 1: S = (Q K^T) * scale, 16 j-tiles per wave.
  {
    int col = lane & 15, hi = (lane >> 4) & 1;
    for (int jt = w * 16; jt < (w + 1) * 16; ++jt) {
      int j0 = jt * 16;
      const ushort_t* kptr =
          qkv + ((size_t)(b * NN + j0) * (3 * HH * DHD) + HH * DHD + h * DHD);
      v16bf bk0 = ld_tile(kptr,      3 * HH * DHD, lane);
      v16bf bk1 = ld_tile(kptr + 32, 3 * HH * DHD, lane);
      v8f c = {};
      c = wmma_bf16(aq0, bk0, c);
      c = wmma_bf16(aq1, bk1, c);
#pragma unroll
      for (int r = 0; r < 8; ++r)
        Sld[(r + hi * 8) * NN + j0 + col] = f2bf(c[r] * scale);
    }
  }
  __syncthreads();

  // Phase 2a: row max (8 partials per row).
  {
    int row = t & 15, sub = t >> 4;
    float mx = -1e30f;
    for (int j = sub * (NN / 8); j < (sub + 1) * (NN / 8); ++j)
      mx = fmaxf(mx, bf2f(Sld[row * NN + j]));
    part[row * 8 + sub] = mx;
  }
  __syncthreads();
  if (t < 16) {
    float mx = -1e30f;
    for (int s = 0; s < 8; ++s) mx = fmaxf(mx, part[t * 8 + s]);
    rowmax[t] = mx;
  }
  __syncthreads();

  // Phase 2b: P = exp(S - max) in place (bf16), accumulate partial sums.
  {
    int row = t & 15, sub = t >> 4;
    float rm = rowmax[row], sm = 0.0f;
    for (int j = sub * (NN / 8); j < (sub + 1) * (NN / 8); ++j) {
      float e = __expf(bf2f(Sld[row * NN + j]) - rm);
      Sld[row * NN + j] = f2bf(e);
      sm += e;
    }
    part[row * 8 + sub] = sm;
  }
  __syncthreads();
  if (t < 16) {
    float s = 0.0f;
    for (int k = 0; k < 8; ++k) s += part[t * 8 + k];
    rowsum[t] = s;
  }
  __syncthreads();

  // Phase 3: O = P V / rowsum; wave w owns d block [w*16, w*16+16).
  {
    int d0 = w * 16;
    int col = lane & 15, hi = (lane >> 4) & 1;
    const ushort_t* vbase =
        qkv + ((size_t)b * NN * (3 * HH * DHD) + 2 * HH * DHD + h * DHD);
    v8f c = {};
    for (int j0 = 0; j0 < NN; j0 += 32) {
      v16bf ap = ld_tile(Sld + j0, NN, lane);       // A tile straight from LDS
      v16bf bv;                                     // V^T gather (strided)
#pragma unroll
      for (int i = 0; i < 16; ++i)
        bv[i] = us2bf(vbase[(size_t)(j0 + kmapB(i, lane)) * (3 * HH * DHD) + d0 + col]);
      c = wmma_bf16(ap, bv, c);
    }
#pragma unroll
    for (int r = 0; r < 8; ++r) {
      int m = r + hi * 8;
      float o = c[r] / rowsum[m];
      obf[((size_t)(b * NN + m0 + m)) * (HH * DHD) + h * DHD + d0 + col] = f2bf(o);
    }
  }
}

// ---------------- SE gate: column mean over tokens (2-stage, deterministic) --
__global__ void k_colmean_part(const float* __restrict__ XR, float* __restrict__ P) {
  int seg = blockIdx.x, b = blockIdx.y, d = threadIdx.x;
  float s = 0.0f;
  for (int n = seg * (NN / 16); n < (seg + 1) * (NN / 16); ++n)
    s += XR[((size_t)b * NN + n) * DD + d];
  P[((size_t)b * 16 + seg) * DD + d] = s;
}
__global__ void k_colmean_fin(const float* __restrict__ P, float* __restrict__ CM) {
  int b = blockIdx.x, d = threadIdx.x;
  float s = 0.0f;
  for (int k = 0; k < 16; ++k) s += P[((size_t)b * 16 + k) * DD + d];
  CM[b * DD + d] = s * (1.0f / NN);
}

// sigmoid(relu(cm @ w1) @ w2), w1: [D, D/4], w2: [D/4, D]
__global__ void k_se(const float* __restrict__ CM, const float* __restrict__ w1,
                     const float* __restrict__ w2, float* __restrict__ S) {
  __shared__ float cm[DD];
  __shared__ float hid[DD / 4];
  int b = blockIdx.x, t = threadIdx.x;
  cm[t] = CM[b * DD + t];
  __syncthreads();
  if (t < DD / 4) {
    float hsum = 0.0f;
    for (int d = 0; d < DD; ++d) hsum += cm[d] * w1[d * (DD / 4) + t];
    hid[t] = fmaxf(hsum, 0.0f);
  }
  __syncthreads();
  float s = 0.0f;
  for (int j = 0; j < DD / 4; ++j) s += hid[j] * w2[j * DD + t];
  S[b * DD + t] = 1.0f / (1.0f + __expf(-s));
}

__global__ void k_scale(const float* __restrict__ XR, const float* __restrict__ S,
                        float* __restrict__ dst) {
  int row = blockIdx.x, t = threadIdx.x, b = row >> 10;
  size_t i = (size_t)row * DD + t;
  dst[i] = XR[i] * S[b * DD + t];
}

// ---------------- ACB depthwise conv (1x3 along N) + center tap + BN + GELU --
__global__ void k_conv(const ushort_t* __restrict__ Y, const float* __restrict__ wh,
                       const float* __restrict__ bh, const float* __restrict__ wv,
                       const float* __restrict__ bv, const float* __restrict__ bng,
                       const float* __restrict__ bnb, const float* __restrict__ bnm,
                       const float* __restrict__ bnv, ushort_t* __restrict__ Z) {
  int n = blockIdx.x, b = blockIdx.y;
  for (int m = threadIdx.x; m < MMF; m += blockDim.x) {
    size_t base = ((size_t)b * NN + n) * MMF + m;
    float y0  = bf2f(Y[base]);
    float ym1 = (n > 0)      ? bf2f(Y[base - MMF]) : 0.0f;
    float yp1 = (n < NN - 1) ? bf2f(Y[base + MMF]) : 0.0f;
    float ch = wh[m * 3 + 0] * ym1 + wh[m * 3 + 1] * y0 + wh[m * 3 + 2] * yp1 + bh[m];
    float cv = wv[m * 3 + 1] * y0 + bv[m];
    float z  = ch + cv;
    z = (z - bnm[m]) * rsqrtf(bnv[m] + EPSV) * bng[m] + bnb[m];
    Z[base] = f2bf(geluf(z));
  }
}

// ---------------------------------------------------------------------------
extern "C" void kernel_launch(void* const* d_in, const int* in_sizes, int n_in,
                              void* d_out, int out_size, void* d_ws, size_t ws_size,
                              hipStream_t stream) {
  (void)in_sizes; (void)n_in; (void)out_size; (void)ws_size;
  const float* x     = (const float*)d_in[0];
  const float* ln1_g = (const float*)d_in[1];
  const float* ln1_b = (const float*)d_in[2];
  const float* w_qkv = (const float*)d_in[3];
  const float* w_out = (const float*)d_in[4];
  const float* b_out = (const float*)d_in[5];
  const float* ln2_g = (const float*)d_in[6];
  const float* ln2_b = (const float*)d_in[7];
  const float* w_fc1 = (const float*)d_in[8];
  const float* b_fc1 = (const float*)d_in[9];
  const float* wh    = (const float*)d_in[10];
  const float* bh    = (const float*)d_in[11];
  const float* wv    = (const float*)d_in[12];
  const float* bv    = (const float*)d_in[13];
  const float* bn_g  = (const float*)d_in[14];
  const float* bn_b  = (const float*)d_in[15];
  const float* bn_m  = (const float*)d_in[16];
  const float* bn_v  = (const float*)d_in[17];
  const float* w_fc2 = (const float*)d_in[18];
  const float* b_fc2 = (const float*)d_in[19];
  const float* ls_w1 = (const float*)d_in[20];
  const float* ls_w2 = (const float*)d_in[21];

  const int rows = BB * NN;           // 8192
  const int Q3   = 3 * HH * DHD;      // 768

  // ws bump allocator (256B aligned)
  char* ws = (char*)d_ws;
  size_t off = 0;
  auto alloc = [&](size_t bytes) -> void* {
    void* p = ws + off;
    off = (off + bytes + 255) & ~(size_t)255;
    return p;
  };
  ushort_t* WqkvT = (ushort_t*)alloc((size_t)LNUM * Q3 * DD * 2);
  ushort_t* WoutT = (ushort_t*)alloc((size_t)LNUM * DD * DD * 2);
  ushort_t* Wfc1T = (ushort_t*)alloc((size_t)LNUM * MMF * DD * 2);
  ushort_t* Wfc2T = (ushort_t*)alloc((size_t)LNUM * DD * MMF * 2);
  ushort_t* Hbf   = (ushort_t*)alloc((size_t)rows * DD * 2);
  ushort_t* QKV   = (ushort_t*)alloc((size_t)rows * Q3 * 2);
  ushort_t* Obf   = (ushort_t*)alloc((size_t)rows * DD * 2);
  ushort_t* Ybf   = (ushort_t*)alloc((size_t)rows * MMF * 2);
  ushort_t* Zbf   = (ushort_t*)alloc((size_t)rows * MMF * 2);
  float*    XR    = (float*)alloc((size_t)rows * DD * 4);
  float*    XA    = (float*)alloc((size_t)rows * DD * 4);
  float*    XB    = (float*)alloc((size_t)rows * DD * 4);
  float*    CMP   = (float*)alloc((size_t)BB * 16 * DD * 4);
  float*    CM    = (float*)alloc((size_t)BB * DD * 4);
  float*    Sg    = (float*)alloc((size_t)BB * DD * 4);

  // weight cast + transpose (tiny)
  for (int l = 0; l < LNUM; ++l) {
    int tot;
    tot = DD * Q3;
    k_castT<<<(tot + 255) / 256, 256, 0, stream>>>(w_qkv + (size_t)l * tot,
        WqkvT + (size_t)l * Q3 * DD, DD, Q3, tot);
    tot = DD * DD;
    k_castT<<<(tot + 255) / 256, 256, 0, stream>>>(w_out + (size_t)l * tot,
        WoutT + (size_t)l * DD * DD, DD, DD, tot);
    tot = DD * MMF;
    k_castT<<<(tot + 255) / 256, 256, 0, stream>>>(w_fc1 + (size_t)l * tot,
        Wfc1T + (size_t)l * MMF * DD, DD, MMF, tot);
    tot = MMF * DD;
    k_castT<<<(tot + 255) / 256, 256, 0, stream>>>(w_fc2 + (size_t)l * tot,
        Wfc2T + (size_t)l * DD * MMF, MMF, DD, tot);
  }

  for (int l = 0; l < LNUM; ++l) {
    const float* Xin = (l == 0) ? x : XB;
    // --- MHSA ---
    k_ln_cast<<<rows, DD, 0, stream>>>(Xin, ln1_g + l * DD, ln1_b + l * DD, Hbf);
    { dim3 g(Q3 / 32, rows / 128);
      k_gemm<<<g, 256, 0, stream>>>(Hbf, WqkvT + (size_t)l * Q3 * DD,
          nullptr, nullptr, nullptr, QKV, DD, Q3, 0); }
    { dim3 g(NN / 16, HH, BB);
      k_attn<<<g, 128, 0, stream>>>(QKV, Obf); }
    { dim3 g(DD / 32, rows / 128);
      k_gemm<<<g, 256, 0, stream>>>(Obf, WoutT + (size_t)l * DD * DD,
          b_out + l * DD, Xin, XR, nullptr, DD, DD, 0); }
    k_colmean_part<<<dim3(16, BB), DD, 0, stream>>>(XR, CMP);
    k_colmean_fin<<<BB, DD, 0, stream>>>(CMP, CM);
    k_se<<<BB, DD, 0, stream>>>(CM, ls_w1 + (size_t)l * DD * (DD / 4),
                                ls_w2 + (size_t)l * (DD / 4) * DD, Sg);
    k_scale<<<rows, DD, 0, stream>>>(XR, Sg, XA);
    // --- MLP + ACB conv ---
    k_ln_cast<<<rows, DD, 0, stream>>>(XA, ln2_g + l * DD, ln2_b + l * DD, Hbf);
    { dim3 g(MMF / 32, rows / 128);
      k_gemm<<<g, 256, 0, stream>>>(Hbf, Wfc1T + (size_t)l * MMF * DD,
          b_fc1 + l * MMF, nullptr, nullptr, Ybf, DD, MMF, 1); }
    { dim3 g(NN, BB);
      k_conv<<<g, 256, 0, stream>>>(Ybf, wh + (size_t)l * MMF * 3, bh + l * MMF,
          wv + (size_t)l * MMF * 3, bv + l * MMF, bn_g + l * MMF, bn_b + l * MMF,
          bn_m + l * MMF, bn_v + l * MMF, Zbf); }
    { dim3 g(DD / 32, rows / 128);
      k_gemm<<<g, 256, 0, stream>>>(Zbf, Wfc2T + (size_t)l * DD * MMF,
          b_fc2 + l * DD, XA, XR, nullptr, MMF, DD, 0); }
    k_colmean_part<<<dim3(16, BB), DD, 0, stream>>>(XR, CMP);
    k_colmean_fin<<<BB, DD, 0, stream>>>(CMP, CM);
    k_se<<<BB, DD, 0, stream>>>(CM, ls_w1 + (size_t)l * DD * (DD / 4),
                                ls_w2 + (size_t)l * (DD / 4) * DD, Sg);
    float* dst = (l == LNUM - 1) ? (float*)d_out : XB;
    k_scale<<<rows, DD, 0, stream>>>(XR, Sg, dst);
  }
}
